// GPT2SelfAttention_23433341567608
// MI455X (gfx1250) — matrix operand order repaired and checked by reading, not compiled, not run on hardware
//
#include <hip/hip_runtime.h>
#include <hip/hip_bf16.h>

// ---------------------------------------------------------------------------
// GPT-2 self-attention forward, CDNA5 (gfx1250), wave32 + v_wmma_f32_16x16x32_bf16
// B=4, S=2048, D=1024, H=16, HD=64
// Pipeline: [GEMM qkv, bf16 WMMA] -> [flash attention, async-LDS + WMMA] ->
//           [GEMM out, bf16 WMMA]
// ---------------------------------------------------------------------------

typedef __attribute__((ext_vector_type(16))) __bf16 v16bf;
typedef __attribute__((ext_vector_type(8)))  __bf16 v8bf;
typedef __attribute__((ext_vector_type(4)))  __bf16 v4bf;
typedef __attribute__((ext_vector_type(8)))  float  v8f;

static __device__ __forceinline__ v16bf frag16(const __bf16* p0, const __bf16* p1) {
    union { v16bf v; v8bf h[2]; } u;
    u.h[0] = *(const v8bf*)p0;
    u.h[1] = *(const v8bf*)p1;
    return u.v;
}

static __device__ __forceinline__ v8f wmma_bf16(v16bf a, v16bf b, v8f c) {
    return __builtin_amdgcn_wmma_f32_16x16x32_bf16(false, a, false, b, (short)0, c,
                                                   false, false);
}

// Low 32 bits of a generic pointer to __shared__ == LDS byte offset (aperture model)
static __device__ __forceinline__ unsigned lds_off(const void* p) {
    return (unsigned)(unsigned long long)p;
}

// Async DMA: 16 bytes/lane global -> LDS, tracked by ASYNCcnt (no VGPR round trip)
static __device__ __forceinline__ void async_b128(unsigned loff, const void* g) {
    asm volatile("global_load_async_to_lds_b128 %0, %1, off"
                 :: "v"(loff), "v"(g) : "memory");
}

#define SEQ    2048
#define DMODEL 1024
#define NH     16
#define HD     64

// ---------------------------------------------------------------------------
// bf16 WMMA GEMM:  C[M,N] = A[M,1024] * Bw[1024,N] (+ bias)
// MODE 0: N=3072, epilogue scatters q*0.125 [B,H,S,64], k [B,H,S,64],
//         v [B,H,64,S] (bf16; softmax scale folded into q)
// MODE 1: N=1024, plain fp32 epilogue into out
// Block tile 128x128, 8 waves (4x2), wave tile 32x64, K-step 32.
// Software-pipelined: next k-tile global loads overlap current tile's WMMAs.
// ---------------------------------------------------------------------------
template <int MODE, int NCOLS>
__global__ __launch_bounds__(256) void gemm_bf16_kernel(
    const float* __restrict__ A, const float* __restrict__ Bw,
    const float* __restrict__ bias,
    __bf16* __restrict__ qws, __bf16* __restrict__ kws, __bf16* __restrict__ vws,
    float* __restrict__ outp)
{
    const int m0   = blockIdx.y * 128;
    const int n0   = blockIdx.x * 128;
    const int tid  = threadIdx.x;
    const int lane = tid & 31;
    const int wave = tid >> 5;
    const int wm   = wave >> 1;      // 0..3  (rows of 32)
    const int wn   = wave & 1;       // 0..1  (cols of 64)
    const int l16  = lane & 15;
    const int lh   = lane >> 4;

    __shared__ __align__(16) __bf16 a_lds[128][40];  // [m][k], stride 80B (=5*16)
    __shared__ __align__(16) __bf16 b_lds[128][40];  // [n][k] (transposed)

    v8f acc[2][4] = {};
    float4 aR[4], bR[4];

    auto gload = [&](int kk) {
#pragma unroll
        for (int it = 0; it < 4; ++it) {
            int ia = tid + it * 256;                 // A: 1024 float4 groups
            int row = ia >> 3, kg = (ia & 7) * 4;
            aR[it] = *(const float4*)(A + (size_t)(m0 + row) * DMODEL + kk + kg);
            int kr = ia >> 5, ng = (ia & 31) * 4;    // B: 1024 float4 groups
            bR[it] = *(const float4*)(Bw + (size_t)(kk + kr) * NCOLS + n0 + ng);
        }
    };
    auto lstore = [&]() {
#pragma unroll
        for (int it = 0; it < 4; ++it) {
            int ia = tid + it * 256;
            int row = ia >> 3, kg = (ia & 7) * 4;
            v4bf w; w[0] = (__bf16)aR[it].x; w[1] = (__bf16)aR[it].y;
            w[2] = (__bf16)aR[it].z;  w[3] = (__bf16)aR[it].w;
            *(v4bf*)&a_lds[row][kg] = w;             // ds_store_b64
            int kr = ia >> 5, ng = (ia & 31) * 4;
            b_lds[ng + 0][kr] = (__bf16)bR[it].x;
            b_lds[ng + 1][kr] = (__bf16)bR[it].y;
            b_lds[ng + 2][kr] = (__bf16)bR[it].z;
            b_lds[ng + 3][kr] = (__bf16)bR[it].w;
        }
    };

    gload(0);
    for (int kk = 0; kk < DMODEL; kk += 32) {
        __syncthreads();                  // previous fragments consumed
        lstore();
        __syncthreads();
        if (kk + 32 < DMODEL) gload(kk + 32);   // overlap HBM latency with WMMAs

        v16bf af[2], bfr[4];
        const int k0a = lh * 16;
        const int k0b = lh * 16;
#pragma unroll
        for (int tm = 0; tm < 2; ++tm) {
            int row = wm * 32 + tm * 16 + l16;
            af[tm] = frag16(&a_lds[row][k0a], &a_lds[row][k0a + 8]);
        }
#pragma unroll
        for (int tn = 0; tn < 4; ++tn) {
            int col = wn * 64 + tn * 16 + l16;
            bfr[tn] = frag16(&b_lds[col][k0b], &b_lds[col][k0b + 8]);
        }
#pragma unroll
        for (int tm = 0; tm < 2; ++tm)
#pragma unroll
            for (int tn = 0; tn < 4; ++tn)
                acc[tm][tn] = wmma_bf16(af[tm], bfr[tn], acc[tm][tn]);
    }

    // ---- epilogue; C layout: lane holds col n (l16), rows i + 8*lh ----
#pragma unroll
    for (int tm = 0; tm < 2; ++tm)
#pragma unroll
        for (int tn = 0; tn < 4; ++tn) {
            const int n  = n0 + wn * 64 + tn * 16 + l16;
            const float bv = bias[n];
            if (MODE == 0) {
                const int h = n / (3 * HD);
                const int r = n % (3 * HD);
                const int t = r / HD;
                const int d = r % HD;
#pragma unroll
                for (int i = 0; i < 8; ++i) {
                    int m = m0 + wm * 32 + tm * 16 + i + lh * 8;
                    int b = m >> 11, s = m & (SEQ - 1);
                    float val = acc[tm][tn][i] + bv;
                    size_t bh = (size_t)(b * NH + h);
                    if (t == 0)                           // fold 1/sqrt(HD) into q
                        qws[(bh * SEQ + s) * HD + d] = (__bf16)(val * 0.125f);
                    else if (t == 1)
                        kws[(bh * SEQ + s) * HD + d] = (__bf16)val;
                    else
                        vws[(bh * HD + d) * SEQ + s] = (__bf16)val;  // V^T
                }
            } else {
#pragma unroll
                for (int i = 0; i < 8; ++i) {
                    int m = m0 + wm * 32 + tm * 16 + i + lh * 8;
                    outp[(size_t)m * NCOLS + n] = acc[tm][tn][i] + bv;
                }
            }
        }
}

// ---------------------------------------------------------------------------
// Flash attention: grid (S/128, B*H), 256 threads = 8 waves, 16 queries/wave.
// K/V double-buffered in LDS via async global->LDS DMA (ASYNCcnt); scores and
// P*V through WMMA; online softmax with wave32 shfl reductions.
// Causal handling: fully-masked key blocks skipped per-wave; per-element mask
// only on diagonal blocks (wave-uniform branch).
// k layout: [bh][s][d] (row = key);  v layout: [bh][d][s] (pre-transposed).
// ---------------------------------------------------------------------------
__global__ __launch_bounds__(256) void flash_attn_kernel(
    const __bf16* __restrict__ q, const __bf16* __restrict__ k,
    const __bf16* __restrict__ v, float* __restrict__ ctx)
{
    const int bh  = blockIdx.y;             // b*16 + h
    const int q0  = blockIdx.x * 128;
    const int tid = threadIdx.x;
    const int lane = tid & 31, wv = tid >> 5;
    const int l16 = lane & 15, lh = lane >> 4;

    const __bf16* qp = q + (size_t)bh * SEQ * HD;
    const __bf16* kp = k + (size_t)bh * SEQ * HD;
    const __bf16* vp = v + (size_t)bh * HD * SEQ;   // [d][s]

    __shared__ __align__(16) __bf16 k_lds[2][64][72];    // [buf][key][d]
    __shared__ __align__(16) __bf16 v_lds[2][64][72];    // [buf][d][key]
    __shared__ __align__(16) __bf16 p_lds[8][16][72];    // per-wave P staging

    // Q fragments (held for whole kernel): rows = l16, d chunks per A-layout
    v16bf qf[2];
    {
        const __bf16* qr = qp + (size_t)(q0 + wv * 16 + l16) * HD;
#pragma unroll
        for (int j = 0; j < 2; ++j) {
            int d1 = j * 32 + lh * 16;
            qf[j] = frag16(qr + d1, qr + d1 + 8);
        }
    }

    // async-issue K/V tile for key-block kb into LDS buffer bu
    auto issue_kv = [&](int kb, int bu) {
        const int kst = kb << 6;
#pragma unroll
        for (int it = 0; it < 2; ++it) {
            int idx = tid + it * 256;             // 512 chunks of 8 bf16
            int row = idx >> 3, c = (idx & 7) * 8;
            async_b128(lds_off(&k_lds[bu][row][c]),
                       kp + (size_t)(kst + row) * HD + c);       // K: [key][d]
            async_b128(lds_off(&v_lds[bu][row][c]),
                       vp + (size_t)row * SEQ + kst + c);        // V^T: [d][key]
        }
    };

    v8f cacc[4] = {};
    float mrow[8], lrow[8];
#pragma unroll
    for (int i = 0; i < 8; ++i) { mrow[i] = -1e30f; lrow[i] = 0.f; }

    const int qlo = q0 + wv * 16;           // wave's smallest query index
    const int nkb = (q0 + 128) >> 6;        // causal: keys < q0+128
    issue_kv(0, 0);
    for (int kb = 0; kb < nkb; ++kb) {
        const int bu = kb & 1;
        asm volatile("s_wait_asynccnt 0" ::: "memory");  // this wave's DMA done
        __syncthreads();                                 // all waves' tiles in LDS
        if (kb + 1 < nkb) issue_kv(kb + 1, bu ^ 1);      // prefetch next block
        const int kst = kb << 6;

        if (kst > qlo + 15) continue;       // block fully masked for this wave
        const bool diag = (kst + 63) > qlo; // per-element mask needed?

        // ---- scores: 4 tiles of 16 keys, 2 WMMAs each (K=64 over d) ----
        v8f sc[4];
        const int qbase = qlo + lh * 8;
#pragma unroll
        for (int t = 0; t < 4; ++t) {
            v8f s = {};
            const int col = t * 16 + l16;   // key within block
#pragma unroll
            for (int j = 0; j < 2; ++j) {
                int d1 = j * 32 + lh * 16;
                v16bf bk = frag16(&k_lds[bu][col][d1], &k_lds[bu][col][d1 + 8]);
                s = wmma_bf16(qf[j], bk, s);
            }
            if (diag) {                      // wave-uniform branch
                const int key = kst + col;
#pragma unroll
                for (int i = 0; i < 8; ++i)
                    if (key > qbase + i) s[i] = -10000.0f;   // matches reference
            }
            sc[t] = s;
        }

        // ---- online softmax (16-lane row reductions via shfl_xor) ----
        float mnew[8], alpha[8], rsum[8];
#pragma unroll
        for (int i = 0; i < 8; ++i) {
            float mx = fmaxf(fmaxf(sc[0][i], sc[1][i]), fmaxf(sc[2][i], sc[3][i]));
#pragma unroll
            for (int off = 1; off < 16; off <<= 1)
                mx = fmaxf(mx, __shfl_xor(mx, off, 32));
            mnew[i]  = fmaxf(mrow[i], mx);
            alpha[i] = __expf(mrow[i] - mnew[i]);
            rsum[i]  = 0.f;
        }
#pragma unroll
        for (int n = 0; n < 4; ++n)
#pragma unroll
            for (int i = 0; i < 8; ++i) cacc[n][i] *= alpha[i];
#pragma unroll
        for (int t = 0; t < 4; ++t)
#pragma unroll
            for (int i = 0; i < 8; ++i) {
                float p = __expf(sc[t][i] - mnew[i]);
                sc[t][i] = p;
                rsum[i] += p;
            }
#pragma unroll
        for (int i = 0; i < 8; ++i) {
            float su = rsum[i];
#pragma unroll
            for (int off = 1; off < 16; off <<= 1) su += __shfl_xor(su, off, 32);
            lrow[i] = lrow[i] * alpha[i] + su;
            mrow[i] = mnew[i];
        }

        // ---- P: C-layout -> LDS -> A-layout (wave-private region) ----
#pragma unroll
        for (int t = 0; t < 4; ++t)
#pragma unroll
            for (int i = 0; i < 8; ++i)
                p_lds[wv][i + lh * 8][t * 16 + l16] = (__bf16)sc[t][i];
        asm volatile("s_wait_dscnt 0" ::: "memory");  // own-wave LDS RAW

        // ---- ctx += P(16x64) x V(64x64): 4 d-tiles x 2 K-steps ----
#pragma unroll
        for (int n = 0; n < 4; ++n) {
#pragma unroll
            for (int j = 0; j < 2; ++j) {
                int d1 = j * 32 + lh * 16;
                v16bf pa = frag16(&p_lds[wv][l16][d1], &p_lds[wv][l16][d1 + 8]);
                int dc = n * 16 + l16;
                int kc = j * 32 + lh * 16;
                v16bf vb = frag16(&v_lds[bu][dc][kc], &v_lds[bu][dc][kc + 8]);
                cacc[n] = wmma_bf16(pa, vb, cacc[n]);
            }
        }
    }

    // ---- normalize & write ctx fp32 [B,S,D] (heads concatenated) ----
    const int b = bh >> 4, h = bh & 15;
#pragma unroll
    for (int n = 0; n < 4; ++n) {
        const int d = h * HD + n * 16 + l16;
#pragma unroll
        for (int i = 0; i < 8; ++i) {
            int qq = q0 + wv * 16 + i + lh * 8;
            ctx[((size_t)(b * SEQ + qq)) * DMODEL + d] = cacc[n][i] / lrow[i];
        }
    }
}

// ---------------------------------------------------------------------------
extern "C" void kernel_launch(void* const* d_in, const int* in_sizes, int n_in,
                              void* d_out, int out_size, void* d_ws, size_t ws_size,
                              hipStream_t stream) {
    (void)in_sizes; (void)n_in; (void)out_size; (void)ws_size;
    const float* x    = (const float*)d_in[0];
    const float* Wqkv = (const float*)d_in[1];
    const float* bqkv = (const float*)d_in[2];
    const float* Wd   = (const float*)d_in[3];
    const float* bd   = (const float*)d_in[4];
    float* out = (float*)d_out;

    char* ws = (char*)d_ws;
    const size_t nBH = (size_t)4 * NH * SEQ * HD;        // 8,388,608 elems
    __bf16* qws = (__bf16*)(ws);                          // 16 MB  [b,h,s,d] (pre-scaled)
    __bf16* kws = (__bf16*)(ws + nBH * 2);                // 16 MB  [b,h,s,d]
    __bf16* vws = (__bf16*)(ws + nBH * 4);                // 16 MB  [b,h,d,s]
    float*  ctx = (float*)(ws + nBH * 6);                 // 33.5 MB

    dim3 blk(256);
    dim3 g1(3 * DMODEL / 128, (4 * SEQ) / 128);           // 24 x 64
    gemm_bf16_kernel<0, 3 * DMODEL><<<g1, blk, 0, stream>>>(
        x, Wqkv, bqkv, qws, kws, vws, nullptr);

    dim3 g2(SEQ / 128, 4 * NH);                           // 16 x 64
    flash_attn_kernel<<<g2, blk, 0, stream>>>(qws, kws, vws, ctx);

    dim3 g3(DMODEL / 128, (4 * SEQ) / 128);               // 8 x 64
    gemm_bf16_kernel<1, DMODEL><<<g3, blk, 0, stream>>>(
        ctx, Wd, bd, nullptr, nullptr, nullptr, out);
}
